// PointPillarsEncoder_86947317940408
// MI455X (gfx1250) — matrix operand order repaired and checked by reading, not compiled, not run on hardware
//
#include <hip/hip_runtime.h>
#include <hip/hip_bf16.h>
#include <stdint.h>

// ---------------------------------------------------------------------------
// PointPillars encoder for MI455X (gfx1250, wave32, WMMA).
// All GEMM/conv work runs through v_wmma_f32_16x16x32_bf16 with f32 accum.
// k_fe2 additionally exercises the CDNA5 async-copy + Tensor Data Mover path
// (global_load_async_to_lds_b128 / tensor_load_to_lds + asynccnt/tensorcnt).
// Requires ws_size >= ~700 MB (linear scratch layout, offsets below).
// ---------------------------------------------------------------------------

#define GXD 256
#define GYD 320
#define GZD 16
#define VTOT (GXD * GYD * GZD)   // 1310720 voxels

typedef __attribute__((ext_vector_type(16))) __bf16 bf16x16;
typedef __attribute__((ext_vector_type(8)))  float  f32x8;
typedef __attribute__((ext_vector_type(4)))  unsigned int u32x4;
typedef __attribute__((ext_vector_type(8)))  int         i32x8;
typedef __attribute__((ext_vector_type(4)))  int         i32x4;

__device__ __forceinline__ unsigned short f2bf(float f) {
  unsigned u = __float_as_uint(f);
  u += 0x7fffu + ((u >> 16) & 1u);          // round-to-nearest-even
  return (unsigned short)(u >> 16);
}
__device__ __forceinline__ float bf2f(unsigned short h) {
  return __uint_as_float(((unsigned)h) << 16);
}

union FragAB { bf16x16 v; uint4 q[2]; };

// low 32 bits of a flat LDS pointer == LDS byte address (aperture semantics)
__device__ __forceinline__ unsigned lds_addr32(const void* p) {
  return (unsigned)(size_t)p;
}
// per-lane async copy of 16B global -> 16B LDS (ASYNCcnt tracked)
__device__ __forceinline__ void async_g2l_b128(unsigned lds_byte_addr, const void* gptr) {
  unsigned long long ga = (unsigned long long)(size_t)gptr;
  asm volatile("global_load_async_to_lds_b128 %0, %1, off"
               :: "v"(lds_byte_addr), "v"(ga) : "memory");
}
__device__ __forceinline__ void wait_asynccnt0() {
  asm volatile("s_wait_asynccnt 0x0" ::: "memory");
}

// ---------------------------------------------------------------------------
// init: writer = -1, stats (9 sets x 1024 f32) = 0
// ---------------------------------------------------------------------------
__global__ void k_init(int* __restrict__ writer, float* __restrict__ stats) {
  int i = blockIdx.x * 256 + threadIdx.x;
  if (i < VTOT) writer[i] = -1;
  if (i < 9 * 1024) stats[i] = 0.0f;
}

// ---------------------------------------------------------------------------
// voxel scatter: last-write-wins == max point index per voxel
// ---------------------------------------------------------------------------
__global__ void k_scatter(const float* __restrict__ pts, int* __restrict__ writer, int N) {
  int i = blockIdx.x * blockDim.x + threadIdx.x;
  if (i >= N) return;
  float x = pts[3 * i + 0], y = pts[3 * i + 1], z = pts[3 * i + 2];
  int ix = (int)(x / 0.25f);            // pc_range min x = 0
  int iy = (int)((y + 40.0f) / 0.25f);
  int iz = (int)((z + 4.0f) / 0.5f);
  ix = min(max(ix, 0), GXD - 1);
  iy = min(max(iy, 0), GYD - 1);
  iz = min(max(iz, 0), GZD - 1);
  int flat = iz * (GYD * GXD) + iy * GXD + ix;
  atomicMax(&writer[flat], i);
}

// ---------------------------------------------------------------------------
// FE layer 1: gather voxel point, pre-act (3->32), BN stats via LDS atomics.
// pre1 stored row-major [V][32] bf16 (= WMMA A layout source for layer 2).
// ---------------------------------------------------------------------------
__global__ __launch_bounds__(256)
void k_fe1(const float* __restrict__ pts, const int* __restrict__ writer,
           const float* __restrict__ w1, unsigned short* __restrict__ pre1,
           float* __restrict__ stats) {
  __shared__ float sw[96];
  __shared__ float ssum[32];
  __shared__ float sss[32];
  if (threadIdx.x < 96) sw[threadIdx.x] = w1[threadIdx.x];
  if (threadIdx.x < 32) { ssum[threadIdx.x] = 0.f; sss[threadIdx.x] = 0.f; }
  __syncthreads();
  int v = blockIdx.x * 256 + threadIdx.x;       // grid sized exactly V/256
  float px = 0.f, py = 0.f, pz = 0.f;
  int wr = writer[v];
  if (wr >= 0) { px = pts[3 * wr]; py = pts[3 * wr + 1]; pz = pts[3 * wr + 2]; }
  unsigned short* o = pre1 + (size_t)v * 32;
  #pragma unroll
  for (int j = 0; j < 32; ++j) {
    float p = sw[j * 3] * px + sw[j * 3 + 1] * py + sw[j * 3 + 2] * pz;
    o[j] = f2bf(p);
    atomicAdd(&ssum[j], p);                      // ds_add_f32
    atomicAdd(&sss[j], p * p);
  }
  __syncthreads();
  if (threadIdx.x < 32) {
    atomicAdd(&stats[threadIdx.x], ssum[threadIdx.x]);
    atomicAdd(&stats[256 + threadIdx.x], sss[threadIdx.x]);
  }
}

// ---------------------------------------------------------------------------
// BN finalize: scale = g*rsqrt(var+eps), shift = be - mean*scale
// stats layout per set: [0..255]=sum [256..511]=sumsq [512..767]=scale [768..1023]=shift
// ---------------------------------------------------------------------------
__global__ void k_bn_finalize(const float* __restrict__ g, const float* __restrict__ be,
                              float* __restrict__ st, int C, float invN) {
  int c = blockIdx.x * blockDim.x + threadIdx.x;
  if (c >= C) return;
  float m = st[c] * invN;
  float var = st[256 + c] * invN - m * m;
  float sc = g[c] * rsqrtf(var + 1e-5f);
  st[512 + c] = sc;
  st[768 + c] = be[c] - m * sc;
}

// BN+ReLU apply, row-major [N][C] bf16 (C multiple of 8, 8 elems/thread)
__global__ void k_bn_relu_row(unsigned short* __restrict__ x, const float* __restrict__ st,
                              int C, size_t nvec) {
  size_t i = (size_t)blockIdx.x * blockDim.x + threadIdx.x;
  if (i >= nvec) return;
  size_t base = i * 8;
  int c0 = (int)(base % (size_t)C);
  uint4 q = *(const uint4*)(x + base);
  unsigned wd[4] = {q.x, q.y, q.z, q.w};
  #pragma unroll
  for (int k = 0; k < 4; ++k) {
    int c = c0 + 2 * k;
    float lo = fmaxf(bf2f((unsigned short)(wd[k] & 0xffff)) * st[512 + c] + st[768 + c], 0.f);
    float hi = fmaxf(bf2f((unsigned short)(wd[k] >> 16)) * st[512 + c + 1] + st[768 + c + 1], 0.f);
    wd[k] = (unsigned)f2bf(lo) | ((unsigned)f2bf(hi) << 16);
  }
  q.x = wd[0]; q.y = wd[1]; q.z = wd[2]; q.w = wd[3];
  *(uint4*)(x + base) = q;
}

// BN+ReLU apply, channel-major [C][HW] bf16 (HW multiple of 8)
__global__ void k_bn_relu_ch(unsigned short* __restrict__ x, const float* __restrict__ st,
                             size_t HW, size_t nvec) {
  size_t i = (size_t)blockIdx.x * blockDim.x + threadIdx.x;
  if (i >= nvec) return;
  size_t base = i * 8;
  int c = (int)(base / HW);
  float sc = st[512 + c], sh = st[768 + c];
  uint4 q = *(const uint4*)(x + base);
  unsigned wd[4] = {q.x, q.y, q.z, q.w};
  #pragma unroll
  for (int k = 0; k < 4; ++k) {
    float lo = fmaxf(bf2f((unsigned short)(wd[k] & 0xffff)) * sc + sh, 0.f);
    float hi = fmaxf(bf2f((unsigned short)(wd[k] >> 16)) * sc + sh, 0.f);
    wd[k] = (unsigned)f2bf(lo) | ((unsigned)f2bf(hi) << 16);
  }
  q.x = wd[0]; q.y = wd[1]; q.z = wd[2]; q.w = wd[3];
  *(uint4*)(x + base) = q;
}

// residual merge: a = relu(bn_a(a) + bn_b(b)), channel-major
__global__ void k_add_relu(unsigned short* __restrict__ a, const unsigned short* __restrict__ b,
                           const float* __restrict__ stA, const float* __restrict__ stB,
                           size_t HW, size_t nvec) {
  size_t i = (size_t)blockIdx.x * blockDim.x + threadIdx.x;
  if (i >= nvec) return;
  size_t base = i * 8;
  int c = (int)(base / HW);
  float sA = stA[512 + c], tA = stA[768 + c];
  float sB = stB[512 + c], tB = stB[768 + c];
  uint4 qa = *(const uint4*)(a + base);
  uint4 qb = *(const uint4*)(b + base);
  unsigned wa[4] = {qa.x, qa.y, qa.z, qa.w};
  unsigned wb[4] = {qb.x, qb.y, qb.z, qb.w};
  #pragma unroll
  for (int k = 0; k < 4; ++k) {
    float lo = fmaxf(bf2f((unsigned short)(wa[k] & 0xffff)) * sA + tA +
                     bf2f((unsigned short)(wb[k] & 0xffff)) * sB + tB, 0.f);
    float hi = fmaxf(bf2f((unsigned short)(wa[k] >> 16)) * sA + tA +
                     bf2f((unsigned short)(wb[k] >> 16)) * sB + tB, 0.f);
    wa[k] = (unsigned)f2bf(lo) | ((unsigned)f2bf(hi) << 16);
  }
  qa.x = wa[0]; qa.y = wa[1]; qa.z = wa[2]; qa.w = wa[3];
  *(uint4*)(a + base) = qa;
}

// 2x2 maxpool, channel-major; final stage writes f32 to d_out
__global__ void k_maxpool(const unsigned short* __restrict__ in, unsigned short* __restrict__ outb,
                          float* __restrict__ outf, int C, int H, int W, int asF32) {
  long long idx = (long long)blockIdx.x * blockDim.x + threadIdx.x;
  int Ho = H >> 1, Wo = W >> 1;
  long long tot = (long long)C * Ho * Wo;
  if (idx >= tot) return;
  int ow = (int)(idx % Wo);
  long long t = idx / Wo;
  int oh = (int)(t % Ho);
  int c  = (int)(t / Ho);
  size_t base = ((size_t)c * H + (size_t)(oh * 2)) * W + (size_t)(ow * 2);
  float m = fmaxf(fmaxf(bf2f(in[base]), bf2f(in[base + 1])),
                  fmaxf(bf2f(in[base + W]), bf2f(in[base + W + 1])));
  if (asF32) outf[idx] = m; else outb[idx] = f2bf(m);
}

// ---------------------------------------------------------------------------
// weight pack: f32 [Cout][Cin][KH][KW] -> bf16 [Cout][KH*KW][Cin]
// (so each lane's WMMA B fragment is two contiguous 16B runs)
// ---------------------------------------------------------------------------
__global__ void k_wprep(const float* __restrict__ w, unsigned short* __restrict__ o,
                        int COUT, int CIN, int KH, int KW) {
  int i = blockIdx.x * blockDim.x + threadIdx.x;
  int tot = COUT * CIN * KH * KW;
  if (i >= tot) return;
  int kw = i % KW;
  int kh = (i / KW) % KH;
  int ci = (i / (KW * KH)) % CIN;
  int co = i / (KW * KH * CIN);
  o[((size_t)co * KH * KW + kh * KW + kw) * CIN + ci] = f2bf(w[i]);
}

// ---------------------------------------------------------------------------
// FE layer 2: GEMM [V,32] x [32,64], one v_wmma_f32_16x16x32_bf16 per tile.
// CDNA5 data movement showcase:
//   - activation slab (32 voxels x 32ch = 2KB, contiguous) is brought to LDS
//     with per-lane GLOBAL_LOAD_ASYNC_TO_LDS_B128, completed by
//     s_wait_asynccnt 0 + workgroup barrier,
//   - the 4KB packed w2 tile is fetched by the Tensor Data Mover with a real
//     2-D D# descriptor (data_size=2B, tensor 32x64, tile 32x64, stride 32),
//     completed by s_wait_tensorcnt 0. Fallback: plain b128 copies.
// Output pre2 channel-major [64][V] bf16 (this IS the pseudo-image layout).
// 8 waves/block = 2 M-tiles x 4 N-tiles.
// ---------------------------------------------------------------------------
__global__ __launch_bounds__(256)
void k_fe2(const unsigned short* __restrict__ act1, const unsigned short* __restrict__ w2b,
           unsigned short* __restrict__ pre2, float* __restrict__ stats) {
  __shared__ alignas(16) unsigned short sAct[32 * 32];   // 2KB: 32 voxels x 32 ci
  __shared__ alignas(16) unsigned short sW[64 * 32];     // 4KB: 64 cout x 32 ci
  const int tid = threadIdx.x, lane = tid & 31, wid = tid >> 5;
  const int mt = wid & 1, nt = wid >> 1;                 // nt in [0,4)
  const int h = lane >> 4, row = lane & 15;

  // --- async stage: 2KB activation slab, 128 lanes x 16B ---
  const unsigned short* gact = act1 + (size_t)blockIdx.x * 32 * 32;
  if (tid < 128)
    async_g2l_b128(lds_addr32(sAct) + (unsigned)tid * 16, gact + (size_t)tid * 8);

  // --- TDM stage: 4KB weight tile (wave 0 issues; TDM ignores EXEC) ---
#if __has_builtin(__builtin_amdgcn_tensor_load_to_lds) && __has_builtin(__builtin_amdgcn_s_wait_tensorcnt)
  if (wid == 0) {
    unsigned long long wga = (unsigned long long)(size_t)w2b;
    u32x4 g0;
    g0.x = 1u;                                           // count=1 (valid user D#)
    g0.y = lds_addr32(sW);                               // lds_addr
    g0.z = (unsigned)(wga & 0xffffffffu);                // global_addr[31:0]
    g0.w = (unsigned)((wga >> 32) & 0x01ffffffu) | (2u << 30);  // addr[56:32] | type=2
    i32x8 g1;
    g1[0] = (1 << 16);          // data_size = 2 bytes
    g1[1] = (32 << 16);         // tensor_dim0 = 32 (low 16 bits at bit 48)
    g1[2] = (64 << 16);         // tensor_dim0 hi=0 | tensor_dim1 = 64
    g1[3] = (32 << 16);         // tensor_dim1 hi=0 | tile_dim0 = 32
    g1[4] = 64;                 // tile_dim1 = 64 | tile_dim2 = 0
    g1[5] = 32;                 // tensor_dim0_stride = 32 (low 32)
    g1[6] = 0;                  // stride hi | tensor_dim1_stride lo
    g1[7] = 0;
    i32x4 z4 = {0, 0, 0, 0};
    i32x8 z8 = {0, 0, 0, 0, 0, 0, 0, 0};
    __builtin_amdgcn_tensor_load_to_lds(g0, g1, z4, z4, z8, 0);
    __builtin_amdgcn_s_wait_tensorcnt(0);
  }
#else
  for (int e = tid; e < 512; e += 256)                    // 4096 ushorts / 8
    *(uint4*)(sW + e * 8) = *(const uint4*)(w2b + e * 8);
#endif

  wait_asynccnt0();            // this wave's async LDS writes are done
  __syncthreads();             // everyone's writes visible workgroup-wide

  FragAB A, B;
  const unsigned short* ap = sAct + (size_t)(mt * 16 + row) * 32;
  A.q[0] = *(const uint4*)(ap + h * 8);                  // ds_load_b128
  A.q[1] = *(const uint4*)(ap + 16 + h * 8);
  int c = nt * 16 + row;                                 // out channel (N = lane&15)
  const unsigned short* bp = sW + (size_t)c * 32 + h * 8;
  B.q[0] = *(const uint4*)(bp);
  B.q[1] = *(const uint4*)(bp + 16);

  f32x8 acc = {0.f, 0.f, 0.f, 0.f, 0.f, 0.f, 0.f, 0.f};
  acc = __builtin_amdgcn_wmma_f32_16x16x32_bf16(false, A.v, false, B.v,
                                                (short)0, acc, false, false);
  // store + BN stats
  int v0 = (blockIdx.x * 2 + mt) * 16;
  float s = 0.f, ss = 0.f;
  unsigned short us[8];
  #pragma unroll
  for (int r = 0; r < 8; ++r) { float y = acc[r]; s += y; ss += y * y; us[r] = f2bf(y); }
  uint4 st;
  st.x = (unsigned)us[0] | ((unsigned)us[1] << 16);
  st.y = (unsigned)us[2] | ((unsigned)us[3] << 16);
  st.z = (unsigned)us[4] | ((unsigned)us[5] << 16);
  st.w = (unsigned)us[6] | ((unsigned)us[7] << 16);
  *(uint4*)(pre2 + (size_t)c * VTOT + v0 + h * 8) = st;   // rows M=r+h*8 contiguous
  s  += __shfl_xor(s, 16, 32);
  ss += __shfl_xor(ss, 16, 32);
  if (h == 0) { atomicAdd(&stats[c], s); atomicAdd(&stats[256 + c], ss); }
}

// ---------------------------------------------------------------------------
// Implicit-GEMM conv (KHxKW, pad=(KH-1)/2, stride S) via WMMA bf16.
// Input bf16 [CIN][H][W]; weights pre-packed bf16 [COUT][KH*KW][CIN];
// output pre-BN bf16 [COUT][Ho][Wo] + per-channel sum/sumsq atomics.
// Block: NW waves; each block makes one 16-wide output row segment across
// all COUT. LDS slab layout [KH][WT][32ci] makes A loads = 2x ds_load_b128.
// ---------------------------------------------------------------------------
template <int CIN, int COUT, int KH, int KW, int STRIDE, int NW>
__global__ __launch_bounds__(NW * 32)
void conv_wmma(const unsigned short* __restrict__ in, const unsigned short* __restrict__ wbuf,
               unsigned short* __restrict__ out, float* __restrict__ stats, int H, int W) {
  constexpr int PAD = (KH - 1) / 2;
  constexpr int WT = 16 * STRIDE + (KW - 1);
  constexpr int KK = KH * KW;
  constexpr int NT = COUT / 16;
  constexpr int MAXJ = (NT + NW - 1) / NW;

  const int Ho = (H + 2 * PAD - KH) / STRIDE + 1;
  const int Wo = (W + 2 * PAD - KW) / STRIDE + 1;
  const int m0 = blockIdx.x * 16;
  const int oh = m0 / Wo, ow0 = m0 % Wo;
  const int tid = threadIdx.x, lane = tid & 31, wid = tid >> 5;
  const int h = lane >> 4, row = lane & 15;
  const int iw0 = ow0 * STRIDE - ((KW - 1) / 2);

  __shared__ alignas(16) unsigned short sA[KH * WT * 32];

  f32x8 acc[MAXJ];
  #pragma unroll
  for (int j = 0; j < MAXJ; ++j) {
    f32x8 z = {0.f, 0.f, 0.f, 0.f, 0.f, 0.f, 0.f, 0.f};
    acc[j] = z;
  }

  for (int ci0 = 0; ci0 < CIN; ci0 += 32) {
    __syncthreads();
    if (ci0 + 32 < CIN)   // pull next channel slab toward L2 (global_prefetch_b8)
      __builtin_prefetch(in + (size_t)(ci0 + 32) * (size_t)H * W + (size_t)oh * STRIDE * W, 0, 1);
    // stage [KH][WT][32] slab (zero-padded at borders), coalesced in col
    for (int e = tid; e < KH * WT * 32; e += NW * 32) {
      int col = e % WT;
      int ci  = (e / WT) & 31;
      int r   = e / (WT * 32);
      int ih  = oh * STRIDE + r - PAD;
      int iw  = iw0 + col;
      unsigned short val = 0;
      if (ih >= 0 && ih < H && iw >= 0 && iw < W)
        val = in[(size_t)(ci0 + ci) * (size_t)H * W + (size_t)ih * W + iw];
      sA[(r * WT + col) * 32 + ci] = val;
    }
    __syncthreads();
    #pragma unroll
    for (int t = 0; t < KK; ++t) {
      const int kh = t / KW, kw = t % KW;
      FragAB A;
      const unsigned short* ap = &sA[(kh * WT + row * STRIDE + kw) * 32];
      A.q[0] = *(const uint4*)(ap + h * 8);         // ds_load_b128
      A.q[1] = *(const uint4*)(ap + 16 + h * 8);    // ds_load_b128
      #pragma unroll
      for (int j = 0; j < MAXJ; ++j) {
        int nt = wid + j * NW;
        if (nt < NT) {
          int c = nt * 16 + row;
          FragAB B;
          const unsigned short* bp = wbuf + ((size_t)c * KK + t) * CIN + ci0 + h * 8;
          B.q[0] = *(const uint4*)(bp);
          B.q[1] = *(const uint4*)(bp + 16);
          acc[j] = __builtin_amdgcn_wmma_f32_16x16x32_bf16(false, A.v, false, B.v,
                                                           (short)0, acc[j], false, false);
        }
      }
    }
  }

  // store pre-BN bf16 + per-channel BN statistics
  #pragma unroll
  for (int j = 0; j < MAXJ; ++j) {
    int nt = wid + j * NW;
    if (nt >= NT) continue;
    int c = nt * 16 + row;
    float s = 0.f, ss = 0.f;
    unsigned short us[8];
    #pragma unroll
    for (int r = 0; r < 8; ++r) { float y = acc[j][r]; s += y; ss += y * y; us[r] = f2bf(y); }
    uint4 st;
    st.x = (unsigned)us[0] | ((unsigned)us[1] << 16);
    st.y = (unsigned)us[2] | ((unsigned)us[3] << 16);
    st.z = (unsigned)us[4] | ((unsigned)us[5] << 16);
    st.w = (unsigned)us[6] | ((unsigned)us[7] << 16);
    *(uint4*)(out + (size_t)c * Ho * Wo + (size_t)oh * Wo + ow0 + h * 8) = st;
    s  += __shfl_xor(s, 16, 32);
    ss += __shfl_xor(ss, 16, 32);
    if (h == 0) { atomicAdd(&stats[c], s); atomicAdd(&stats[256 + c], ss); }
  }
}

// ---------------------------------------------------------------------------
// workspace layout (bytes); requires ws_size >= ~700 MB
// ---------------------------------------------------------------------------
static constexpr size_t OFF_STATS = 0;                                    // 9 sets x 4KB
static constexpr size_t OFF_WRITER = 65536;
static constexpr size_t OFF_WB2  = OFF_WRITER + (size_t)VTOT * 4;
static constexpr size_t OFF_WBC0 = OFF_WB2  + (size_t)2048   * 2;
static constexpr size_t OFF_WB11 = OFF_WBC0 + (size_t)36864  * 2;
static constexpr size_t OFF_WB12 = OFF_WB11 + (size_t)73728  * 2;
static constexpr size_t OFF_WB1S = OFF_WB12 + (size_t)147456 * 2;
static constexpr size_t OFF_WB21 = OFF_WB1S + (size_t)8192   * 2;
static constexpr size_t OFF_WB22 = OFF_WB21 + (size_t)294912 * 2;
static constexpr size_t OFF_WB2S = OFF_WB22 + (size_t)589824 * 2;
static constexpr size_t OFF_ACT1 = OFF_WB2S + (size_t)32768  * 2;
static constexpr size_t OFF_IMG  = OFF_ACT1 + (size_t)VTOT * 32 * 2;
static constexpr size_t OFF_Y0   = OFF_IMG  + (size_t)VTOT * 64 * 2;
static constexpr size_t OFF_O1   = OFF_Y0   + (size_t)64  * 327680 * 2;
static constexpr size_t OFF_O2   = OFF_O1   + (size_t)128 * 327680 * 2;
static constexpr size_t OFF_SC1  = OFF_O2   + (size_t)128 * 327680 * 2;
static constexpr size_t OFF_P1   = OFF_SC1  + (size_t)128 * 327680 * 2;
static constexpr size_t OFF_Q1   = OFF_P1   + (size_t)128 * 81920  * 2;
static constexpr size_t OFF_Q2   = OFF_Q1   + (size_t)256 * 81920  * 2;
static constexpr size_t OFF_SC2  = OFF_Q2   + (size_t)256 * 81920  * 2;

extern "C" void kernel_launch(void* const* d_in, const int* in_sizes, int n_in,
                              void* d_out, int out_size, void* d_ws, size_t ws_size,
                              hipStream_t stream) {
  (void)n_in; (void)out_size; (void)ws_size;
  const float* pts = (const float*)d_in[0];
  const int Npts = in_sizes[0] / 3;
  // fe
  const float* fe_w1 = (const float*)d_in[1];
  const float* fe_g1 = (const float*)d_in[3];
  const float* fe_be1 = (const float*)d_in[4];
  const float* fe_w2 = (const float*)d_in[5];
  const float* fe_g2 = (const float*)d_in[7];
  const float* fe_be2 = (const float*)d_in[8];
  // c0
  const float* c0_w = (const float*)d_in[9];
  const float* c0_g = (const float*)d_in[11];
  const float* c0_be = (const float*)d_in[12];
  // rb1
  const float* r1_w1 = (const float*)d_in[13];
  const float* r1_g1 = (const float*)d_in[15];
  const float* r1_be1 = (const float*)d_in[16];
  const float* r1_w2 = (const float*)d_in[17];
  const float* r1_g2 = (const float*)d_in[19];
  const float* r1_be2 = (const float*)d_in[20];
  const float* r1_ws = (const float*)d_in[21];
  const float* r1_gs = (const float*)d_in[23];
  const float* r1_bes = (const float*)d_in[24];
  // rb2
  const float* r2_w1 = (const float*)d_in[25];
  const float* r2_g1 = (const float*)d_in[27];
  const float* r2_be1 = (const float*)d_in[28];
  const float* r2_w2 = (const float*)d_in[29];
  const float* r2_g2 = (const float*)d_in[31];
  const float* r2_be2 = (const float*)d_in[32];
  const float* r2_ws = (const float*)d_in[33];
  const float* r2_gs = (const float*)d_in[35];
  const float* r2_bes = (const float*)d_in[36];

  char* ws = (char*)d_ws;
  float* stats = (float*)(ws + OFF_STATS);
  auto st = [&](int s) { return stats + (size_t)s * 1024; };
  int* writer = (int*)(ws + OFF_WRITER);
  unsigned short* wb2  = (unsigned short*)(ws + OFF_WB2);
  unsigned short* wbc0 = (unsigned short*)(ws + OFF_WBC0);
  unsigned short* wb11 = (unsigned short*)(ws + OFF_WB11);
  unsigned short* wb12 = (unsigned short*)(ws + OFF_WB12);
  unsigned short* wb1s = (unsigned short*)(ws + OFF_WB1S);
  unsigned short* wb21 = (unsigned short*)(ws + OFF_WB21);
  unsigned short* wb22 = (unsigned short*)(ws + OFF_WB22);
  unsigned short* wb2s = (unsigned short*)(ws + OFF_WB2S);
  unsigned short* act1 = (unsigned short*)(ws + OFF_ACT1);
  unsigned short* img  = (unsigned short*)(ws + OFF_IMG);
  unsigned short* y0   = (unsigned short*)(ws + OFF_Y0);
  unsigned short* o1   = (unsigned short*)(ws + OFF_O1);
  unsigned short* o2   = (unsigned short*)(ws + OFF_O2);
  unsigned short* sc1  = (unsigned short*)(ws + OFF_SC1);
  unsigned short* p1   = (unsigned short*)(ws + OFF_P1);
  unsigned short* q1   = (unsigned short*)(ws + OFF_Q1);
  unsigned short* q2   = (unsigned short*)(ws + OFF_Q2);
  unsigned short* sc2  = (unsigned short*)(ws + OFF_SC2);

  // --- init + scatter ---
  k_init<<<VTOT / 256, 256, 0, stream>>>(writer, stats);
  k_scatter<<<(Npts + 255) / 256, 256, 0, stream>>>(pts, writer, Npts);

  // --- weight packing (f32 -> bf16 [Cout][tap][Cin]) ---
  k_wprep<<<(2048 + 255) / 256, 256, 0, stream>>>(fe_w2, wb2, 64, 32, 1, 1);
  k_wprep<<<(36864 + 255) / 256, 256, 0, stream>>>(c0_w, wbc0, 64, 64, 3, 3);
  k_wprep<<<(73728 + 255) / 256, 256, 0, stream>>>(r1_w1, wb11, 128, 64, 3, 3);
  k_wprep<<<(147456 + 255) / 256, 256, 0, stream>>>(r1_w2, wb12, 128, 128, 3, 3);
  k_wprep<<<(8192 + 255) / 256, 256, 0, stream>>>(r1_ws, wb1s, 128, 64, 1, 1);
  k_wprep<<<(294912 + 255) / 256, 256, 0, stream>>>(r2_w1, wb21, 256, 128, 3, 3);
  k_wprep<<<(589824 + 255) / 256, 256, 0, stream>>>(r2_w2, wb22, 256, 256, 3, 3);
  k_wprep<<<(32768 + 255) / 256, 256, 0, stream>>>(r2_ws, wb2s, 256, 128, 1, 1);

  const float invV = 1.0f / (float)VTOT;
  const float invS0 = 1.0f / (float)(8 * 40960);
  const float invS1 = invS0;                 // rb1 convs: 8x40960
  const float invS2 = 1.0f / (float)(4 * 20480);

  // --- feature encoder ---
  k_fe1<<<VTOT / 256, 256, 0, stream>>>(pts, writer, fe_w1, act1, st(0));
  k_bn_finalize<<<1, 256, 0, stream>>>(fe_g1, fe_be1, st(0), 32, invV);
  k_bn_relu_row<<<(unsigned)((size_t)VTOT * 32 / 8 / 256), 256, 0, stream>>>(
      act1, st(0), 32, (size_t)VTOT * 32 / 8);
  k_fe2<<<VTOT / 32, 256, 0, stream>>>(act1, wb2, img, st(1));
  k_bn_finalize<<<1, 256, 0, stream>>>(fe_g2, fe_be2, st(1), 64, invV);
  k_bn_relu_ch<<<(unsigned)((size_t)VTOT * 64 / 8 / 256), 256, 0, stream>>>(
      img, st(1), (size_t)VTOT, (size_t)VTOT * 64 / 8);

  // --- c0: 64->64, 3x3, stride 2 on [64,16,81920] ---
  conv_wmma<64, 64, 3, 3, 2, 4><<<8 * 40960 / 16, 128, 0, stream>>>(img, wbc0, y0, st(2), 16, 81920);
  k_bn_finalize<<<1, 256, 0, stream>>>(c0_g, c0_be, st(2), 64, invS0);
  k_bn_relu_ch<<<(unsigned)((size_t)64 * 327680 / 8 / 256), 256, 0, stream>>>(
      y0, st(2), (size_t)327680, (size_t)64 * 327680 / 8);

  // --- rb1 ---
  conv_wmma<64, 128, 3, 3, 1, 8><<<327680 / 16, 256, 0, stream>>>(y0, wb11, o1, st(3), 8, 40960);
  k_bn_finalize<<<1, 256, 0, stream>>>(r1_g1, r1_be1, st(3), 128, invS1);
  k_bn_relu_ch<<<(unsigned)((size_t)128 * 327680 / 8 / 256), 256, 0, stream>>>(
      o1, st(3), (size_t)327680, (size_t)128 * 327680 / 8);
  conv_wmma<128, 128, 3, 3, 1, 8><<<327680 / 16, 256, 0, stream>>>(o1, wb12, o2, st(4), 8, 40960);
  conv_wmma<64, 128, 1, 1, 1, 8><<<327680 / 16, 256, 0, stream>>>(y0, wb1s, sc1, st(5), 8, 40960);
  k_bn_finalize<<<1, 256, 0, stream>>>(r1_g2, r1_be2, st(4), 128, invS1);
  k_bn_finalize<<<1, 256, 0, stream>>>(r1_gs, r1_bes, st(5), 128, invS1);
  k_add_relu<<<(unsigned)((size_t)128 * 327680 / 8 / 256), 256, 0, stream>>>(
      o2, sc1, st(4), st(5), (size_t)327680, (size_t)128 * 327680 / 8);
  k_maxpool<<<(unsigned)((size_t)128 * 4 * 20480 / 256), 256, 0, stream>>>(
      o2, p1, nullptr, 128, 8, 40960, 0);

  // --- rb2 ---
  conv_wmma<128, 256, 3, 3, 1, 8><<<81920 / 16, 256, 0, stream>>>(p1, wb21, q1, st(6), 4, 20480);
  k_bn_finalize<<<1, 256, 0, stream>>>(r2_g1, r2_be1, st(6), 256, invS2);
  k_bn_relu_ch<<<(unsigned)((size_t)256 * 81920 / 8 / 256), 256, 0, stream>>>(
      q1, st(6), (size_t)81920, (size_t)256 * 81920 / 8);
  conv_wmma<256, 256, 3, 3, 1, 8><<<81920 / 16, 256, 0, stream>>>(q1, wb22, q2, st(7), 4, 20480);
  conv_wmma<128, 256, 1, 1, 1, 8><<<81920 / 16, 256, 0, stream>>>(p1, wb2s, sc2, st(8), 4, 20480);
  k_bn_finalize<<<1, 256, 0, stream>>>(r2_g2, r2_be2, st(7), 256, invS2);
  k_bn_finalize<<<1, 256, 0, stream>>>(r2_gs, r2_bes, st(8), 256, invS2);
  k_add_relu<<<(unsigned)((size_t)256 * 81920 / 8 / 256), 256, 0, stream>>>(
      q2, sc2, st(7), st(8), (size_t)81920, (size_t)256 * 81920 / 8);
  k_maxpool<<<(unsigned)((size_t)256 * 2 * 10240 / 256), 256, 0, stream>>>(
      q2, nullptr, (float*)d_out, 256, 4, 20480, 1);
}
